// Attention_38130719654026
// MI455X (gfx1250) — matrix operand order repaired and checked by reading, not compiled, not run on hardware
//
#include <hip/hip_runtime.h>
#include <hip/hip_bf16.h>
#include <math.h>
#include <stdint.h>

// ---------------------------------------------------------------------------
// Reference collapse (verified algebraically):
//   * top_k result is unused; kv_sel is a broadcast -> all topk logits equal
//     -> softmax uniform -> attn @ v_sel == v. grad (256MB) is never read.
//   * Remaining work:  v = x @ W_qkv[:,1024:1536]          (8192x512x512)
//                      z = permute(gelu(v))                (bijective shuffle)
//                      out = z @ W_out + b_out             (8192x512x512)
//   Permutation: src (p_src,h_src), j = p_src*8 + h_src, dst P=j%16, H=j/16,
//   channel d preserved:  z[b,P,n,H*64+d] = gelu(v[b,p_src,n,h_src*64+d]).
//
// Pipeline: one-shot f16 prep (x -> f16, weights -> transposed f16), then two
// v_wmma_f32_16x16x32_f16 GEMMs whose LDS tiles are filled with the CDNA5
// async global->LDS path (ASYNCcnt) when the builtin is available.
// ---------------------------------------------------------------------------

typedef __attribute__((ext_vector_type(16))) _Float16 v16h;
typedef __attribute__((ext_vector_type(8)))  _Float16 v8h;
typedef __attribute__((ext_vector_type(8)))  float    v8f;
typedef __attribute__((ext_vector_type(4)))  float    f32x4;
typedef int v4i __attribute__((vector_size(16)));   // matches builtin param type

#define BM 128
#define BN 128
#define BK 32
#define LDK 40          // padded LDS row (halves) = 20 dwords -> conflict-free b128 frag loads
#define KTOT (512 / BK) // 16 K-steps

#if __has_builtin(__builtin_amdgcn_global_load_async_to_lds_b128)
#define USE_ASYNC_LDS 1
#else
#define USE_ASYNC_LDS 0
#endif

// Typed addrspace casts via integer round-trip:
// LDS generic pointers carry the LDS offset in the low 32 bits; global generic
// pointers equal their addrspace(1) value.
#define GLOBAL_PTR_B128(p) ((__attribute__((address_space(1))) v4i*)(uintptr_t)(p))
#define LOCAL_PTR_B128(p)  ((__attribute__((address_space(3))) v4i*)(uint32_t)(uintptr_t)(p))

__device__ __forceinline__ void wait_async_lds() {
#if USE_ASYNC_LDS
#if __has_builtin(__builtin_amdgcn_s_wait_asynccnt)
    __builtin_amdgcn_s_wait_asynccnt(0);
#else
    asm volatile("s_wait_asynccnt 0" ::: "memory");
#endif
#endif
}

__device__ __forceinline__ float gelu_exact(float x) {
    return 0.5f * x * (1.0f + erff(x * 0.70710678118654752f));
}

// ISA 16-bit A/B fragment: halves 0..7 and 8..15 are two contiguous 16B runs.
__device__ __forceinline__ v16h pack16(v8h lo, v8h hi) {
    v16h r;
#pragma unroll
    for (int i = 0; i < 8; ++i) { r[i] = lo[i]; r[8 + i] = hi[i]; }
    return r;
}

// ===========================================================================
// Prep: x (f32) -> xh (f16), contiguous
// ===========================================================================
__global__ __launch_bounds__(256) void k_cvt_x(const float* __restrict__ x,
                                               _Float16* __restrict__ xh) {
    const int i = (blockIdx.x * 256 + threadIdx.x) * 8;
    f32x4 a = *(const f32x4*)&x[i];
    f32x4 b = *(const f32x4*)&x[i + 4];
    v8h h;
#pragma unroll
    for (int j = 0; j < 4; ++j) { h[j] = (_Float16)a[j]; h[4 + j] = (_Float16)b[j]; }
    *(v8h*)&xh[i] = h;
}

// Prep: WvT[n][k] = f16(W_qkv[k][1024+n]),  WoutT[n][k] = f16(W_out[k][n])
__global__ __launch_bounds__(256) void k_build_wT(const float* __restrict__ Wqkv,
                                                  const float* __restrict__ Wout,
                                                  _Float16* __restrict__ WvT,
                                                  _Float16* __restrict__ WoutT) {
    const int n = blockIdx.x;
#pragma unroll
    for (int p = 0; p < 2; ++p) {
        const int k = threadIdx.x + p * 256;
        WvT[n * 512 + k]   = (_Float16)Wqkv[(size_t)k * 1536 + 1024 + n];
        WoutT[n * 512 + k] = (_Float16)Wout[(size_t)k * 512 + n];
    }
}

// ===========================================================================
// Unified WMMA GEMM:  C = A[8192x512] @ BT[512x512]^T
//   MODE 0: C -> gelu + permute, f16 store to zOut
//   MODE 1: C -> + bias, f32 store to fOut
// ===========================================================================
template <int MODE>
__global__ __launch_bounds__(256) void gemm_wmma(
    const _Float16* __restrict__ A,   // [8192][512] f16, K contiguous
    const _Float16* __restrict__ BT,  // [512][512]  f16, BT[n][k]
    const float* __restrict__ bias,   // [512] (MODE 1)
    float* __restrict__ fOut,         // [8192][512] (MODE 1)
    _Float16* __restrict__ zOut)      // [8192][512] permuted (MODE 0)
{
    __shared__ __attribute__((aligned(16))) _Float16 As[2][BM][LDK];
    __shared__ __attribute__((aligned(16))) _Float16 Bs[2][BN][LDK];

    const int tid  = threadIdx.x;
    const int lane = tid & 31;
    const int wave = tid >> 5;
    const int wm   = wave & 3;   // wave M offset = wm*32 (2 m-tiles)
    const int wn   = wave >> 2;  // wave N offset = wn*64 (4 n-tiles)
    const int m16  = lane & 15;
    const int g    = lane >> 4;

    const int rowBase = blockIdx.x * BM;
    const int colBase = blockIdx.y * BN;

    // Tile fill: 128 rows x 32 halves (64B) each for A and B.
    // 256 threads x (2 rows x 16B) per matrix; fully coalesced 16B chunks.
    const int c_r = tid >> 2;        // 0..63 (+64 second pass)
    const int c_k = (tid & 3) * 8;   // half offset 0,8,16,24

    auto fill_tiles = [&](int kt, int buf) {
        const int k0 = kt * BK;
#pragma unroll
        for (int p = 0; p < 2; ++p) {
            const int r = c_r + p * 64;
            const _Float16* ga = &A[(size_t)(rowBase + r) * 512 + k0 + c_k];
            const _Float16* gb = &BT[(size_t)(colBase + r) * 512 + k0 + c_k];
            _Float16* la = &As[buf][r][c_k];
            _Float16* lb = &Bs[buf][r][c_k];
#if USE_ASYNC_LDS
            __builtin_amdgcn_global_load_async_to_lds_b128(
                GLOBAL_PTR_B128(ga), LOCAL_PTR_B128(la), 0, 0);
            __builtin_amdgcn_global_load_async_to_lds_b128(
                GLOBAL_PTR_B128(gb), LOCAL_PTR_B128(lb), 0, 0);
#else
            *(v8h*)la = *(const v8h*)ga;
            *(v8h*)lb = *(const v8h*)gb;
#endif
        }
    };

    v8f acc[2][4];
#pragma unroll
    for (int mi = 0; mi < 2; ++mi)
#pragma unroll
        for (int ni = 0; ni < 4; ++ni) acc[mi][ni] = (v8f){};

    fill_tiles(0, 0);

    for (int kt = 0; kt < KTOT; ++kt) {
        wait_async_lds();       // own async fills landed in LDS
        __syncthreads();        // everyone's fills visible; prev reads of nxt done
        const int cur = kt & 1;
        if (kt + 1 < KTOT) fill_tiles(kt + 1, cur ^ 1);

        v16h afrag[2], bfrag[4];
#pragma unroll
        for (int mi = 0; mi < 2; ++mi) {
            const int row = wm * 32 + mi * 16 + m16;
            v8h lo = *(const v8h*)&As[cur][row][g * 8];
            v8h hi = *(const v8h*)&As[cur][row][16 + g * 8];
            afrag[mi] = pack16(lo, hi);
        }
#pragma unroll
        for (int ni = 0; ni < 4; ++ni) {
            const int col = wn * 64 + ni * 16 + m16;
            v8h lo = *(const v8h*)&Bs[cur][col][g * 8];
            v8h hi = *(const v8h*)&Bs[cur][col][16 + g * 8];
            bfrag[ni] = pack16(lo, hi);
        }
#pragma unroll
        for (int mi = 0; mi < 2; ++mi)
#pragma unroll
            for (int ni = 0; ni < 4; ++ni)
                acc[mi][ni] = __builtin_amdgcn_wmma_f32_16x16x32_f16(
                    false, afrag[mi], false, bfrag[ni], (short)0, acc[mi][ni], false, false);
    }

    // Epilogue. C layout: lane n = L%16, VGPR rr -> m = rr + 8*(L/16).
#pragma unroll
    for (int mi = 0; mi < 2; ++mi)
#pragma unroll
        for (int ni = 0; ni < 4; ++ni)
#pragma unroll
            for (int rr = 0; rr < 8; ++rr) {
                const int r = rowBase + wm * 32 + mi * 16 + rr + 8 * g;
                const int c = colBase + wn * 64 + ni * 16 + m16;
                if (MODE == 0) {
                    const float val = gelu_exact(acc[mi][ni][rr]);
                    const int n_pos = r & 255;
                    const int bp    = r >> 8;
                    const int b_i   = bp >> 4;
                    const int p_src = bp & 15;
                    const int h_src = c >> 6;
                    const int d     = c & 63;
                    const int j     = p_src * 8 + h_src;
                    const int P     = j & 15;
                    const int Hh    = j >> 4;
                    zOut[(size_t)((b_i * 16 + P) * 256 + n_pos) * 512 + Hh * 64 + d] =
                        (_Float16)val;
                } else {
                    fOut[(size_t)r * 512 + c] = acc[mi][ni][rr] + bias[c];
                }
            }
}

// ===========================================================================
extern "C" void kernel_launch(void* const* d_in, const int* in_sizes, int n_in,
                              void* d_out, int out_size, void* d_ws, size_t ws_size,
                              hipStream_t stream) {
    const float* x    = (const float*)d_in[0];
    // d_in[1] = grad: mathematically dead in the reference (top_k unused,
    // broadcast kv -> uniform softmax -> attn output == v). Never read.
    const float* Wqkv = (const float*)d_in[2];
    const float* Wout = (const float*)d_in[3];
    const float* bout = (const float*)d_in[4];
    float* out = (float*)d_out;

    // Workspace layout (17 MiB total, all 256B-aligned offsets):
    char* ws = (char*)d_ws;
    _Float16* xh    = (_Float16*)(ws);                               // 8 MiB
    _Float16* WvT   = (_Float16*)(ws + (8u << 20));                  // 512 KiB
    _Float16* WoutT = (_Float16*)(ws + (8u << 20) + (512u << 10));   // 512 KiB
    _Float16* z     = (_Float16*)(ws + (9u << 20));                  // 8 MiB

    dim3 block(256);

    // Prep: f16 conversion + weight transposes (tiny, one-shot)
    k_cvt_x<<<dim3((8192 * 512) / (256 * 8)), block, 0, stream>>>(x, xh);
    k_build_wT<<<dim3(512), block, 0, stream>>>(Wqkv, Wout, WvT, WoutT);

    dim3 grid(8192 / BM, 512 / BN);  // 64 x 4
    gemm_wmma<0><<<grid, block, 0, stream>>>(xh, WvT, nullptr, nullptr, z);
    gemm_wmma<1><<<grid, block, 0, stream>>>(z, WoutT, bout, out, nullptr);
}
